// GraphClassifier_49916109914436
// MI455X (gfx1250) — compile-verified, matrix operand order checked by reading, not run on hardware
//
#include <hip/hip_runtime.h>
#include <hip/hip_bf16.h>

#define N_NODES   100000
#define N_EDGES   1600000
#define N_GRAPHS  512
#define NODE_EMB  64
#define IDENT_EMB 32
#define IN_DIM    102
#define HID       128
#define NUM_CATS  10
#define KPAD      128   // padded inner dim for both layers

typedef __attribute__((ext_vector_type(16))) _Float16 v16h;
typedef __attribute__((ext_vector_type(8)))  _Float16 v8h;
typedef __attribute__((ext_vector_type(8)))  float    v8f;
typedef __attribute__((ext_vector_type(4)))  int      v4i;

// AS-qualified int4 for the async-to-LDS builtin (param types revealed by clang:
// arg0 = v4i __device__* (AS1), arg1 = v4i __shared__* (AS3))
typedef __attribute__((address_space(1))) v4i gv4i;
typedef __attribute__((address_space(3))) v4i lv4i;

#if defined(__has_builtin)
#if __has_builtin(__builtin_amdgcn_global_load_async_to_lds_b128)
#define HAVE_ASYNC_LDS 1
#endif
#if __has_builtin(__builtin_amdgcn_s_wait_asynccnt)
#define HAVE_WAIT_ASYNC 1
#endif
#if __has_builtin(__builtin_amdgcn_rcpf)
#define FAST_RCP(x) __builtin_amdgcn_rcpf(x)
#endif
#endif
#ifndef FAST_RCP
#define FAST_RCP(x) (1.0f / (x))
#endif

// ---------------------------------------------------------------------------
// depth max (for depth normalization)
// ---------------------------------------------------------------------------
__global__ void k_depth_max(const int* __restrict__ depth, int* __restrict__ dmax) {
    int i = blockIdx.x * blockDim.x + threadIdx.x;
    if (i < N_NODES) atomicMax(dmax, depth[i]);
}

// ---------------------------------------------------------------------------
// build x[N, 128] = [emb_node(64) | emb_ident(32) | flags(5) | depth_norm(1) | 0...]
// ---------------------------------------------------------------------------
__global__ void k_build_x(const int* __restrict__ types, const int* __restrict__ idents,
                          const float* __restrict__ flags, const int* __restrict__ depth,
                          const float* __restrict__ emb_node, const float* __restrict__ emb_ident,
                          const int* __restrict__ dmax, float* __restrict__ X) {
    int n = blockIdx.x * blockDim.x + threadIdx.x;
    if (n >= N_NODES) return;
    float inv_dmax = FAST_RCP(fmaxf((float)(*dmax), 1.0f));
    float* xr = X + (size_t)n * KPAD;
    const float* en = emb_node + (size_t)types[n] * NODE_EMB;
    #pragma unroll
    for (int i = 0; i < NODE_EMB; i += 4) *(float4*)(xr + i) = *(const float4*)(en + i);
    const float* ei = emb_ident + (size_t)idents[n] * IDENT_EMB;
    #pragma unroll
    for (int i = 0; i < IDENT_EMB; i += 4) *(float4*)(xr + NODE_EMB + i) = *(const float4*)(ei + i);
    const float* fl = flags + (size_t)n * 5;
    #pragma unroll
    for (int i = 0; i < 5; i++) xr[96 + i] = fl[i];
    xr[101] = (float)depth[n] * inv_dmax;
    #pragma unroll
    for (int i = IN_DIM; i < KPAD; i++) xr[i] = 0.0f;
}

// ---------------------------------------------------------------------------
// weight prep: four W matrices -> f16, transposed to [n][k] (k zero-padded),
// plus fused biases b0 = b_self0+b_neigh0, b1 = b_self1+b_neigh1
// ---------------------------------------------------------------------------
__global__ void k_prep_w(const float* __restrict__ Ws0, const float* __restrict__ Wn0,
                         const float* __restrict__ Ws1, const float* __restrict__ Wn1,
                         const float* __restrict__ bs0, const float* __restrict__ bn0,
                         const float* __restrict__ bs1, const float* __restrict__ bn1,
                         _Float16* __restrict__ Wp, float* __restrict__ b0, float* __restrict__ b1) {
    int tid = blockIdx.x * blockDim.x + threadIdx.x;
    const int WELEMS = 4 * KPAD * HID;
    if (tid < WELEMS) {
        int mat = tid >> 14;          // / (128*128)
        int e   = tid & 16383;
        int n   = e >> 7;             // output column
        int k   = e & 127;            // inner dim
        const float* W; int ksrc;
        switch (mat) {
            case 0:  W = Ws0; ksrc = IN_DIM; break;
            case 1:  W = Wn0; ksrc = IN_DIM; break;
            case 2:  W = Ws1; ksrc = HID;    break;
            default: W = Wn1; ksrc = HID;    break;
        }
        float v = (k < ksrc) ? W[(size_t)k * HID + n] : 0.0f;
        Wp[(size_t)mat * (KPAD * HID) + (size_t)n * KPAD + k] = (_Float16)v;
    } else if (tid < WELEMS + HID) {
        int n = tid - WELEMS;
        b0[n] = bs0[n] + bn0[n];
    } else if (tid < WELEMS + 2 * HID) {
        int n = tid - WELEMS - HID;
        b1[n] = bs1[n] + bn1[n];
    }
}

// ---------------------------------------------------------------------------
// in-degree (over dst)
// ---------------------------------------------------------------------------
__global__ void k_degree(const int* __restrict__ dst, float* __restrict__ deg) {
    int e = blockIdx.x * blockDim.x + threadIdx.x;
    if (e < N_EDGES) atomicAdd(&deg[dst[e]], 1.0f);
}

// ---------------------------------------------------------------------------
// edge scatter-add: agg[dst] += x[src] ; one wave per edge, lane = 4 channels
// ---------------------------------------------------------------------------
__global__ void k_edge_agg(const int* __restrict__ src, const int* __restrict__ dst,
                           const float* __restrict__ X, float* __restrict__ AGG) {
    int wid  = (blockIdx.x * blockDim.x + threadIdx.x) >> 5;
    int lane = threadIdx.x & 31;
    if (wid >= N_EDGES) return;
    int s = src[wid], d = dst[wid];
    float4 v = ((const float4*)(X + (size_t)s * KPAD))[lane];
    float* ad = AGG + (size_t)d * KPAD + lane * 4;
    atomicAdd(ad + 0, v.x);
    atomicAdd(ad + 1, v.y);
    atomicAdd(ad + 2, v.z);
    atomicAdd(ad + 3, v.w);
}

// ---------------------------------------------------------------------------
// WMMA fragment helpers (A: 16x32 f16 layout per ISA 7.12.2; f32 src + scale)
// lanes 0-15 (M=lane):  halves [K0..K7, K16..K23]
// lanes 16-31 (M=lane-16): halves [K8..K15, K24..K31]  (caller passes p at K-base)
// ---------------------------------------------------------------------------
__device__ __forceinline__ v16h load_a_frag(const float* p, float scale) {
    float4 x0 = ((const float4*)p)[0];
    float4 x1 = ((const float4*)p)[1];
    float4 x2 = ((const float4*)p)[4];   // +16 floats
    float4 x3 = ((const float4*)p)[5];
    v16h a;
    a[0]  = (_Float16)(x0.x * scale); a[1]  = (_Float16)(x0.y * scale);
    a[2]  = (_Float16)(x0.z * scale); a[3]  = (_Float16)(x0.w * scale);
    a[4]  = (_Float16)(x1.x * scale); a[5]  = (_Float16)(x1.y * scale);
    a[6]  = (_Float16)(x1.z * scale); a[7]  = (_Float16)(x1.w * scale);
    a[8]  = (_Float16)(x2.x * scale); a[9]  = (_Float16)(x2.y * scale);
    a[10] = (_Float16)(x2.z * scale); a[11] = (_Float16)(x2.w * scale);
    a[12] = (_Float16)(x3.x * scale); a[13] = (_Float16)(x3.y * scale);
    a[14] = (_Float16)(x3.z * scale); a[15] = (_Float16)(x3.w * scale);
    return a;
}

__device__ __forceinline__ v16h load_b_frag(const _Float16* p) {
    v8h lo = ((const v8h*)p)[0];   // K+0..7
    v8h hi = ((const v8h*)p)[2];   // K+16..23
    v16h b;
    #pragma unroll
    for (int i = 0; i < 8; i++) { b[i] = lo[i]; b[i + 8] = hi[i]; }
    return b;
}

// ---------------------------------------------------------------------------
// fused SAGE layer: Y = relu(X @ Ws + (AGG * inv_deg) @ Wn + bias)
// Weights staged to LDS once per block (async-to-LDS path when available),
// then one wave = 16 M-rows x 128 N-cols (8 accumulators), K loop of 4 x 32.
// ---------------------------------------------------------------------------
__global__ void __launch_bounds__(256)
k_sage_gemm(const float* __restrict__ X, const float* __restrict__ AGG,
            const float* __restrict__ deg,
            const _Float16* __restrict__ Ws, const _Float16* __restrict__ Wn,
            const float* __restrict__ bias, float* __restrict__ Y) {
    __shared__ _Float16 lds_w[2 * KPAD * HID];          // 64 KB: [Ws | Wn]
    const int WM = KPAD * HID;                          // 16384 halves per matrix

    int tid = threadIdx.x;
    // stage both weight matrices: 256 threads x 8 x 16B chunks per matrix
#if defined(HAVE_ASYNC_LDS)
    #pragma unroll
    for (int c = 0; c < 8; c++) {
        int h = (tid + c * 256) * 8;                    // half index (16B chunk)
        __builtin_amdgcn_global_load_async_to_lds_b128(
            (gv4i*)(void*)(Ws + h), (lv4i*)&lds_w[h], 0, 0);
        __builtin_amdgcn_global_load_async_to_lds_b128(
            (gv4i*)(void*)(Wn + h), (lv4i*)&lds_w[WM + h], 0, 0);
    }
#if defined(HAVE_WAIT_ASYNC)
    __builtin_amdgcn_s_wait_asynccnt(0);
#else
    asm volatile("s_wait_asynccnt 0x0" ::: "memory");
#endif
#else
    #pragma unroll
    for (int c = 0; c < 8; c++) {
        int h = (tid + c * 256) * 8;
        *(v8h*)&lds_w[h]      = *(const v8h*)(Ws + h);
        *(v8h*)&lds_w[WM + h] = *(const v8h*)(Wn + h);
    }
#endif
    __syncthreads();

    int wave = (blockIdx.x * blockDim.x + threadIdx.x) >> 5;
    int lane = threadIdx.x & 31;
    int m0   = wave * 16;

    if (m0 < N_NODES) {                                  // wave-uniform; EXEC full inside
        int row  = m0 + (lane & 15);
        int koff = (lane < 16) ? 0 : 8;
        float idg = FAST_RCP(fmaxf(deg[row], 1.0f));

        const _Float16* lws = lds_w;
        const _Float16* lwn = lds_w + WM;

        v8f acc[8] = {};
        #pragma unroll
        for (int ks = 0; ks < KPAD; ks += 32) {
            v16h afrag = load_a_frag(X   + (size_t)row * KPAD + ks + koff, 1.0f);
            v16h nfrag = load_a_frag(AGG + (size_t)row * KPAD + ks + koff, idg);
            #pragma unroll
            for (int j = 0; j < 8; j++) {
                int n = j * 16 + (lane & 15);
                v16h bs = load_b_frag(lws + (size_t)n * KPAD + ks + koff);
                v16h bn = load_b_frag(lwn + (size_t)n * KPAD + ks + koff);
                acc[j] = __builtin_amdgcn_wmma_f32_16x16x32_f16(
                             false, afrag, false, bs, (short)0, acc[j], false, false);
                acc[j] = __builtin_amdgcn_wmma_f32_16x16x32_f16(
                             false, nfrag, false, bn, (short)0, acc[j], false, false);
            }
        }

        // D layout: VGPR r -> M = r (+8 for lanes 16-31), N = lane & 15 in tile
        int rbase = m0 + ((lane >= 16) ? 8 : 0);
        int col   = lane & 15;
        #pragma unroll
        for (int j = 0; j < 8; j++) {
            int n = j * 16 + col;
            float b = bias[n];
            #pragma unroll
            for (int r = 0; r < 8; r++) {
                float v = acc[j][r] + b;
                Y[(size_t)(rbase + r) * KPAD + n] = fmaxf(v, 0.0f);
            }
        }
    }
}

// ---------------------------------------------------------------------------
// global mean pool (accumulate): pooled[g] += h[n], counts[g] += 1
// ---------------------------------------------------------------------------
__global__ void k_pool(const float* __restrict__ H, const int* __restrict__ batch,
                       float* __restrict__ pooled, float* __restrict__ counts) {
    int wid  = (blockIdx.x * blockDim.x + threadIdx.x) >> 5;
    int lane = threadIdx.x & 31;
    if (wid >= N_NODES) return;
    int g = batch[wid];
    float4 v = ((const float4*)(H + (size_t)wid * KPAD))[lane];
    float* pg = pooled + (size_t)g * HID + lane * 4;
    atomicAdd(pg + 0, v.x);
    atomicAdd(pg + 1, v.y);
    atomicAdd(pg + 2, v.z);
    atomicAdd(pg + 3, v.w);
    if (lane == 0) atomicAdd(&counts[g], 1.0f);
}

// ---------------------------------------------------------------------------
// heads: g = pooled/counts ; risk = g@W_risk+b ; cat = g@W_cat+b
// out layout: [0,512) risk ; [512, 512+512*10) cats
// ---------------------------------------------------------------------------
__global__ void k_head(const float* __restrict__ pooled, const float* __restrict__ counts,
                       const float* __restrict__ W_risk, const float* __restrict__ b_risk,
                       const float* __restrict__ W_cat,  const float* __restrict__ b_cat,
                       float* __restrict__ out) {
    __shared__ float gv[HID];
    int g = blockIdx.x;
    int t = threadIdx.x;
    float cnt = fmaxf(counts[g], 1.0f);
    gv[t] = pooled[(size_t)g * HID + t] / cnt;
    __syncthreads();
    if (t == 0) {
        float s = b_risk[0];
        #pragma unroll 8
        for (int i = 0; i < HID; i++) s += gv[i] * W_risk[i];
        out[g] = s;
    } else if (t <= NUM_CATS) {
        int c = t - 1;
        float s = b_cat[c];
        #pragma unroll 8
        for (int i = 0; i < HID; i++) s += gv[i] * W_cat[i * NUM_CATS + c];
        out[N_GRAPHS + (size_t)g * NUM_CATS + c] = s;
    }
}

// ---------------------------------------------------------------------------
extern "C" void kernel_launch(void* const* d_in, const int* in_sizes, int n_in,
                              void* d_out, int out_size, void* d_ws, size_t ws_size,
                              hipStream_t stream) {
    const int*   node_type = (const int*)  d_in[0];
    const int*   ident     = (const int*)  d_in[1];
    const float* flags     = (const float*)d_in[2];
    const int*   depth     = (const int*)  d_in[3];
    const int*   edge      = (const int*)  d_in[4];   // [2, E]
    const int*   batch     = (const int*)  d_in[5];
    const float* emb_node  = (const float*)d_in[6];
    const float* emb_ident = (const float*)d_in[7];
    const float* W_self0   = (const float*)d_in[8];
    const float* b_self0   = (const float*)d_in[9];
    const float* W_neigh0  = (const float*)d_in[10];
    const float* b_neigh0  = (const float*)d_in[11];
    const float* W_self1   = (const float*)d_in[12];
    const float* b_self1   = (const float*)d_in[13];
    const float* W_neigh1  = (const float*)d_in[14];
    const float* b_neigh1  = (const float*)d_in[15];
    const float* W_risk    = (const float*)d_in[16];
    const float* b_risk    = (const float*)d_in[17];
    const float* W_cat     = (const float*)d_in[18];
    const float* b_cat     = (const float*)d_in[19];

    const int* src = edge;
    const int* dst = edge + N_EDGES;

    // workspace layout (256B aligned)
    char* ws = (char*)d_ws;
    size_t off = 0;
    auto take = [&](size_t bytes) { size_t o = off; off = (off + bytes + 255) & ~(size_t)255; return o; };
    const size_t XB = (size_t)N_NODES * KPAD * sizeof(float);
    size_t oX      = take(XB);                       // features, later reused for h2
    size_t oAGG    = take(XB);                       // edge aggregation (both layers)
    size_t oH1     = take(XB);                       // layer-0 output
    size_t oDeg    = take((size_t)N_NODES * sizeof(float));
    size_t oWp     = take((size_t)4 * KPAD * HID * sizeof(_Float16));
    size_t oB0     = take(HID * sizeof(float));
    size_t oB1     = take(HID * sizeof(float));
    size_t oPooled = take((size_t)N_GRAPHS * HID * sizeof(float));
    size_t oCnt    = take(N_GRAPHS * sizeof(float));
    size_t oDmax   = take(sizeof(int));

    float*    X      = (float*)(ws + oX);
    float*    AGG    = (float*)(ws + oAGG);
    float*    H1     = (float*)(ws + oH1);
    float*    degp   = (float*)(ws + oDeg);
    _Float16* Wp     = (_Float16*)(ws + oWp);
    float*    b0     = (float*)(ws + oB0);
    float*    b1     = (float*)(ws + oB1);
    float*    pooled = (float*)(ws + oPooled);
    float*    counts = (float*)(ws + oCnt);
    int*      dmax   = (int*)(ws + oDmax);
    float*    out    = (float*)d_out;

    // zero accumulators (capture-safe async memsets, re-run every replay)
    (void)hipMemsetAsync(degp,   0, (size_t)N_NODES * sizeof(float), stream);
    (void)hipMemsetAsync(AGG,    0, XB, stream);
    (void)hipMemsetAsync(pooled, 0, (size_t)N_GRAPHS * HID * sizeof(float), stream);
    (void)hipMemsetAsync(counts, 0, N_GRAPHS * sizeof(float), stream);
    (void)hipMemsetAsync(dmax,   0, sizeof(int), stream);

    const int nodeBlocks = (N_NODES + 255) / 256;
    const int edgeBlocks = (N_EDGES + 255) / 256;

    k_depth_max<<<nodeBlocks, 256, 0, stream>>>(depth, dmax);
    k_build_x<<<nodeBlocks, 256, 0, stream>>>(node_type, ident, flags, depth,
                                              emb_node, emb_ident, dmax, X);
    k_degree<<<edgeBlocks, 256, 0, stream>>>(dst, degp);

    const int prepThreads = 4 * KPAD * HID + 2 * HID;
    k_prep_w<<<(prepThreads + 255) / 256, 256, 0, stream>>>(
        W_self0, W_neigh0, W_self1, W_neigh1,
        b_self0, b_neigh0, b_self1, b_neigh1, Wp, b0, b1);

    // layer 0
    const int edgeWaveBlocks = (N_EDGES * 32 + 255) / 256;        // wave per edge
    k_edge_agg<<<edgeWaveBlocks, 256, 0, stream>>>(src, dst, X, AGG);

    const int mTiles = N_NODES / 16;                               // 6250 exact
    const int gemmBlocks = (mTiles + 7) / 8;                       // 8 waves/block
    k_sage_gemm<<<gemmBlocks, 256, 0, stream>>>(X, AGG, degp,
                                                Wp + 0 * KPAD * HID, Wp + 1 * KPAD * HID,
                                                b0, H1);

    // layer 1 (reuse AGG after re-zero; reuse X buffer for h2)
    (void)hipMemsetAsync(AGG, 0, XB, stream);
    k_edge_agg<<<edgeWaveBlocks, 256, 0, stream>>>(src, dst, H1, AGG);
    float* H2 = X;
    k_sage_gemm<<<gemmBlocks, 256, 0, stream>>>(H1, AGG, degp,
                                                Wp + 2 * KPAD * HID, Wp + 3 * KPAD * HID,
                                                b1, H2);

    // pool + heads
    const int poolWaveBlocks = (N_NODES * 32 + 255) / 256;
    k_pool<<<poolWaveBlocks, 256, 0, stream>>>(H2, batch, pooled, counts);
    k_head<<<N_GRAPHS, HID, 0, stream>>>(pooled, counts, W_risk, b_risk, W_cat, b_cat, out);

    (void)in_sizes; (void)n_in; (void)out_size; (void)ws_size;
}
